// MultiHeadAttention_78134045049371
// MI455X (gfx1250) — matrix operand order repaired and checked by reading, not compiled, not run on hardware
//
#include <hip/hip_runtime.h>

// ---------------------------------------------------------------------------
// MI455X (gfx1250, wave32) fused MHA block.
// All matmuls via v_wmma_f32_16x16x32_bf16 (fp32 accumulate).
// B=2, L=2048, D=1024, H=16, DK=DV=64.  ~77 GFLOP, ~100MB resident (L2-fits)
// -> matrix-core bound. bf16 WMMA, LDS-staged tiles, flash softmax,
//    async global->LDS staging (ASYNCcnt) for the bf16 tiles.
// ---------------------------------------------------------------------------

typedef __attribute__((ext_vector_type(16))) __bf16 bf16x16;
typedef __attribute__((ext_vector_type(8)))  float  f32x8;

// Hardware fptrunc f32->bf16 (RNE); lets the backend use v_cvt bf16 forms.
static __device__ __forceinline__ unsigned short f32_to_bf16(float f) {
    union { __bf16 b; unsigned short u; } c;
    c.b = (__bf16)f;
    return c.u;
}

union Frag16 { uint4 v[2]; bf16x16 f; };

// A-frag (16x32 bf16): lane l -> row (l&15); element i -> k = g*16 + i
// (g = l>>4) ==> one contiguous 16-elem run => 2 x ds_load_b128.
static __device__ __forceinline__ bf16x16 load_a_frag(const unsigned short* row, int g) {
    Frag16 fr;
    fr.v[0] = *reinterpret_cast<const uint4*>(row + g * 16);
    fr.v[1] = *reinterpret_cast<const uint4*>(row + g * 16 + 8);
    return fr.f;
}
// B-frag (32x16 bf16): lane l -> col (l&15); element i -> k = g*16 + i
// ==> one contiguous 16-elem run => 2 x ds_load_b128.
static __device__ __forceinline__ bf16x16 load_b_frag(const unsigned short* row, int g) {
    const uint4* p = reinterpret_cast<const uint4*>(row + g * 16);
    Frag16 fr;
    fr.v[0] = p[0];
    fr.v[1] = p[1];
    return fr.f;
}

// Async global->LDS copy of 64B (row segment); INST_OFFSET advances both the
// global and LDS address, so one LDS base VGPR serves all four b128 ops.
static __device__ __forceinline__ void async_copy_64B(const void* gsrc, void* lds) {
    unsigned ldsa = (unsigned)(size_t)lds;           // low 32 bits = LDS offset
    asm volatile(
        "global_load_async_to_lds_b128 %0, %1, off\n\t"
        "global_load_async_to_lds_b128 %0, %1, off offset:16\n\t"
        "global_load_async_to_lds_b128 %0, %1, off offset:32\n\t"
        "global_load_async_to_lds_b128 %0, %1, off offset:48"
        :: "v"(ldsa), "v"(gsrc) : "memory");
}
static __device__ __forceinline__ void async_copy_32B(const void* gsrc, void* lds) {
    unsigned ldsa = (unsigned)(size_t)lds;
    asm volatile(
        "global_load_async_to_lds_b128 %0, %1, off\n\t"
        "global_load_async_to_lds_b128 %0, %1, off offset:16"
        :: "v"(ldsa), "v"(gsrc) : "memory");
}
static __device__ __forceinline__ void wait_async() {
    asm volatile("s_wait_asynccnt 0x0" ::: "memory");
}

// ---------------------------------------------------------------------------
// Projection GEMM:  Y[4096,1024](bf16) = X[4096,1024](f32) * W[1024,1024]^T + b
// Block: 128 thr (4 waves), tile 64(M) x 128(N); wave w -> cols w*32..w*32+31.
// Per k-step per wave: 8 WMMA vs 12 ds_b128 (A-frags reused across 2 N-frags).
// ---------------------------------------------------------------------------
__global__ __launch_bounds__(128) void proj_gemm_kernel(
    const float* __restrict__ X, const float* __restrict__ W,
    const float* __restrict__ bias, unsigned short* __restrict__ Y)
{
    constexpr int K = 1024, N = 1024, LDT = 40;
    __shared__ __align__(16) unsigned short As[64 * LDT];
    __shared__ __align__(16) unsigned short Bs[128 * LDT];  // W tile, [n][k]

    const int tid = threadIdx.x, lane = tid & 31, wave = tid >> 5;
    const int g = lane >> 4, ln = lane & 15;
    const int m0 = blockIdx.x * 64, n0 = blockIdx.y * 128;
    const int row = tid >> 1, half = tid & 1;

    f32x8 acc[4][2] = {};

    for (int k0 = 0; k0 < K; k0 += 32) {
        // stage A 64x32 (f32->bf16): thread t -> row t/2, 16 elems
        {
            const float4* sa = reinterpret_cast<const float4*>(X + (size_t)(m0 + row) * K + k0 + half * 16);
            union { uint4 u[2]; unsigned short s[16]; } da;
#pragma unroll
            for (int c = 0; c < 4; ++c) {
                float4 fa = sa[c];
                da.s[c*4+0] = f32_to_bf16(fa.x); da.s[c*4+1] = f32_to_bf16(fa.y);
                da.s[c*4+2] = f32_to_bf16(fa.z); da.s[c*4+3] = f32_to_bf16(fa.w);
            }
            uint4* dA = reinterpret_cast<uint4*>(As + row * LDT + half * 16);
            dA[0] = da.u[0]; dA[1] = da.u[1];
        }
        // stage W 128x32 (f32->bf16): thread t -> full row t (32 elems)
        {
            const float4* sb = reinterpret_cast<const float4*>(W + (size_t)(n0 + tid) * K + k0);
            union { uint4 u[4]; unsigned short s[32]; } db;
#pragma unroll
            for (int c = 0; c < 8; ++c) {
                float4 fb = sb[c];
                db.s[c*4+0] = f32_to_bf16(fb.x); db.s[c*4+1] = f32_to_bf16(fb.y);
                db.s[c*4+2] = f32_to_bf16(fb.z); db.s[c*4+3] = f32_to_bf16(fb.w);
            }
            uint4* dB = reinterpret_cast<uint4*>(Bs + tid * LDT);
            dB[0] = db.u[0]; dB[1] = db.u[1]; dB[2] = db.u[2]; dB[3] = db.u[3];
        }
        __syncthreads();

        bf16x16 bfr0 = load_b_frag(Bs + (wave * 32 + ln) * LDT, g);
        bf16x16 bfr1 = load_b_frag(Bs + (wave * 32 + 16 + ln) * LDT, g);
#pragma unroll
        for (int mt = 0; mt < 4; ++mt) {
            bf16x16 afr = load_a_frag(As + (mt * 16 + ln) * LDT, g);
            acc[mt][0] = __builtin_amdgcn_wmma_f32_16x16x32_bf16(
                false, afr, false, bfr0, (short)0, acc[mt][0], false, false);
            acc[mt][1] = __builtin_amdgcn_wmma_f32_16x16x32_bf16(
                false, afr, false, bfr1, (short)0, acc[mt][1], false, false);
        }
        __syncthreads();
    }

#pragma unroll
    for (int nf = 0; nf < 2; ++nf) {
        const int n = n0 + wave * 32 + nf * 16 + ln;
        const float bn = bias[n];
#pragma unroll
        for (int mt = 0; mt < 4; ++mt)
#pragma unroll
            for (int r = 0; r < 8; ++r) {
                int m = m0 + mt * 16 + r + 8 * g;        // C-frag: VGPR r -> row r+8g
                Y[(size_t)m * N + n] = f32_to_bf16(acc[mt][nf][r] + bn);
            }
    }
}

// ---------------------------------------------------------------------------
// Flash attention over bf16 heads: per block one (b,h) and 64 query rows.
// Wave w owns 16 query rows, full DV=64 accum. Q/K tiles staged by
// GLOBAL_LOAD_ASYNC_TO_LDS_B128 (ASYNCcnt); V is transpose-scattered.
// ---------------------------------------------------------------------------
__global__ __launch_bounds__(128) void attn_kernel(
    const unsigned short* __restrict__ Qh, const unsigned short* __restrict__ Kh,
    const unsigned short* __restrict__ Vh, unsigned short* __restrict__ Oout)
{
    constexpr int L = 2048, H = 16, DH = 64, NH = H * DH;  // NH = 1024
    constexpr int LDK = 72;
    __shared__ __align__(16) unsigned short Kt[64 * LDK];   // [key][dk] (stages Q first)
    __shared__ __align__(16) unsigned short Vts[64 * LDK];  // [dv][key]  (transposed)
    __shared__ __align__(16) unsigned short Ps[64 * LDK];   // [qrow][key] (C->A relayout)

    const int tid = threadIdx.x, lane = tid & 31, wave = tid >> 5;
    const int g = lane >> 4, ln = lane & 15;
    const int q0 = blockIdx.x * 64;
    const int bh = blockIdx.y, b = bh / H, h = bh % H;
    const size_t base = (size_t)b * L * NH + (size_t)h * DH;
    const int row = tid >> 1, half = tid & 1;
    constexpr float inv_temp = 0.125f;                      // 1/sqrt(64)

    // ---- stage Q tile (reuse Kt) via async copy, pull per-wave A-frags
    async_copy_64B(Qh + base + (size_t)(q0 + row) * NH + half * 32,
                   Kt + row * LDK + half * 32);
    wait_async();
    __syncthreads();
    bf16x16 aq[2];
#pragma unroll
    for (int kf = 0; kf < 2; ++kf)
        aq[kf] = load_a_frag(Kt + (wave * 16 + ln) * LDK + kf * 32, g);
    __syncthreads();

    f32x8 o_acc[4] = {};
    float m_r[8], l_r[8];
#pragma unroll
    for (int r = 0; r < 8; ++r) { m_r[r] = -1e30f; l_r[r] = 0.f; }

    for (int kt0 = 0; kt0 < L; kt0 += 64) {
        // ---- stage K [key][dk] async; V^T [dv][key] via register transpose
        async_copy_64B(Kh + base + (size_t)(kt0 + row) * NH + half * 32,
                       Kt + row * LDK + half * 32);
        {
            const uint4* sv = reinterpret_cast<const uint4*>(Vh + base + (size_t)(kt0 + row) * NH + half * 32);
            union { uint4 u[4]; unsigned short s[32]; } dv;
            dv.u[0] = sv[0]; dv.u[1] = sv[1]; dv.u[2] = sv[2]; dv.u[3] = sv[3];
#pragma unroll
            for (int c = 0; c < 32; ++c)
                Vts[(half * 32 + c) * LDK + row] = dv.s[c];  // transpose scatter
        }
        wait_async();
        __syncthreads();

        // ---- S = (Q K^T) * inv_temp   (B[k][n] = K[key=n][dk=k], contiguous)
        f32x8 s[4];
#pragma unroll
        for (int nf = 0; nf < 4; ++nf) {
            f32x8 acc = {};
#pragma unroll
            for (int kf = 0; kf < 2; ++kf) {
                bf16x16 bfr = load_b_frag(Kt + (nf * 16 + ln) * LDK + kf * 32, g);
                acc = __builtin_amdgcn_wmma_f32_16x16x32_bf16(
                    false, aq[kf], false, bfr, (short)0, acc, false, false);
            }
#pragma unroll
            for (int r = 0; r < 8; ++r) acc[r] *= inv_temp;
            s[nf] = acc;
        }

        // ---- online softmax stats per row (row = r + 8g; xor-reduce 16 lanes)
        float mnew[8], alpha[8], rsum[8];
#pragma unroll
        for (int r = 0; r < 8; ++r) {
            float pm = fmaxf(fmaxf(s[0][r], s[1][r]), fmaxf(s[2][r], s[3][r]));
#pragma unroll
            for (int off = 1; off < 16; off <<= 1)
                pm = fmaxf(pm, __shfl_xor(pm, off, 32));
            mnew[r]  = fmaxf(m_r[r], pm);
            alpha[r] = __expf(m_r[r] - mnew[r]);
            rsum[r]  = 0.f;
        }
        // exponentiate, spill P (C-layout) to LDS row-major for A-frag reload
#pragma unroll
        for (int nf = 0; nf < 4; ++nf)
#pragma unroll
            for (int r = 0; r < 8; ++r) {
                float p = __expf(s[nf][r] - mnew[r]);
                rsum[r] += p;
                Ps[(wave * 16 + r + 8 * g) * LDK + nf * 16 + ln] = f32_to_bf16(p);
            }
#pragma unroll
        for (int r = 0; r < 8; ++r) {
            float t = rsum[r];
#pragma unroll
            for (int off = 1; off < 16; off <<= 1)
                t += __shfl_xor(t, off, 32);
            l_r[r] = l_r[r] * alpha[r] + t;
            m_r[r] = mnew[r];
        }
#pragma unroll
        for (int nf = 0; nf < 4; ++nf)
#pragma unroll
            for (int r = 0; r < 8; ++r)
                o_acc[nf][r] *= alpha[r];

        __syncthreads();

        // ---- O += P * V   (B[k][n] = V^T[dv=n][key=k], contiguous in LDS)
#pragma unroll
        for (int kf = 0; kf < 2; ++kf) {
            bf16x16 ap = load_a_frag(Ps + (wave * 16 + ln) * LDK + kf * 32, g);
#pragma unroll
            for (int nf = 0; nf < 4; ++nf) {
                bf16x16 bv = load_b_frag(Vts + (nf * 16 + ln) * LDK + kf * 32, g);
                o_acc[nf] = __builtin_amdgcn_wmma_f32_16x16x32_bf16(
                    false, ap, false, bv, (short)0, o_acc[nf], false, false);
            }
        }
        __syncthreads();
    }

    // ---- normalize and store bf16 attention output [B, L, H*DV]
#pragma unroll
    for (int r = 0; r < 8; ++r) {
        float inv = 1.f / l_r[r];
        int qrow = q0 + wave * 16 + r + 8 * g;
#pragma unroll
        for (int nf = 0; nf < 4; ++nf)
            Oout[base + (size_t)qrow * NH + nf * 16 + ln] = f32_to_bf16(o_acc[nf][r] * inv);
    }
}

// ---------------------------------------------------------------------------
// Gated output: out = sigmoid(O Wg^T + bg) * tanh(O Wfc^T + bfc), fp32 out.
// Tile 64(M) x 128(N), dual-B sharing the A tile (A staged async, bf16).
// Per k-step per wave: 16 WMMA vs 16 ds_b128.
// ---------------------------------------------------------------------------
__global__ __launch_bounds__(128) void out_gemm_kernel(
    const unsigned short* __restrict__ Oin, const float* __restrict__ Wfc,
    const float* __restrict__ bfc, const float* __restrict__ Wg,
    const float* __restrict__ bg, float* __restrict__ out)
{
    constexpr int K = 1024, N = 1024, LDT = 40;
    __shared__ __align__(16) unsigned short As[64 * LDT];
    __shared__ __align__(16) unsigned short Bf[128 * LDT];
    __shared__ __align__(16) unsigned short Bg[128 * LDT];

    const int tid = threadIdx.x, lane = tid & 31, wave = tid >> 5;
    const int g = lane >> 4, ln = lane & 15;
    const int m0 = blockIdx.x * 64, n0 = blockIdx.y * 128;
    const int row = tid >> 1, half = tid & 1;

    f32x8 accf[4][2] = {}, accg[4][2] = {};

    for (int k0 = 0; k0 < K; k0 += 32) {
        // A tile already bf16 -> async copy straight into LDS
        async_copy_32B(Oin + (size_t)(m0 + row) * K + k0 + half * 16,
                       As + row * LDT + half * 16);
        // weight tiles 128x32, f32 -> bf16 in-flight
        {
            const float4* sf = reinterpret_cast<const float4*>(Wfc + (size_t)(n0 + tid) * K + k0);
            const float4* sg = reinterpret_cast<const float4*>(Wg  + (size_t)(n0 + tid) * K + k0);
            union { uint4 u[4]; unsigned short s[32]; } df, dg;
#pragma unroll
            for (int c = 0; c < 8; ++c) {
                float4 ff = sf[c], fg = sg[c];
                df.s[c*4+0] = f32_to_bf16(ff.x); df.s[c*4+1] = f32_to_bf16(ff.y);
                df.s[c*4+2] = f32_to_bf16(ff.z); df.s[c*4+3] = f32_to_bf16(ff.w);
                dg.s[c*4+0] = f32_to_bf16(fg.x); dg.s[c*4+1] = f32_to_bf16(fg.y);
                dg.s[c*4+2] = f32_to_bf16(fg.z); dg.s[c*4+3] = f32_to_bf16(fg.w);
            }
            uint4* dF = reinterpret_cast<uint4*>(Bf + tid * LDT);
            uint4* dG = reinterpret_cast<uint4*>(Bg + tid * LDT);
            dF[0] = df.u[0]; dF[1] = df.u[1]; dF[2] = df.u[2]; dF[3] = df.u[3];
            dG[0] = dg.u[0]; dG[1] = dg.u[1]; dG[2] = dg.u[2]; dG[3] = dg.u[3];
        }
        wait_async();
        __syncthreads();

        bf16x16 bfr[2], bgr[2];
#pragma unroll
        for (int nf = 0; nf < 2; ++nf) {
            bfr[nf] = load_b_frag(Bf + (wave * 32 + nf * 16 + ln) * LDT, g);
            bgr[nf] = load_b_frag(Bg + (wave * 32 + nf * 16 + ln) * LDT, g);
        }
#pragma unroll
        for (int mt = 0; mt < 4; ++mt) {
            bf16x16 afr = load_a_frag(As + (mt * 16 + ln) * LDT, g);
#pragma unroll
            for (int nf = 0; nf < 2; ++nf) {
                accf[mt][nf] = __builtin_amdgcn_wmma_f32_16x16x32_bf16(
                    false, afr, false, bfr[nf], (short)0, accf[mt][nf], false, false);
                accg[mt][nf] = __builtin_amdgcn_wmma_f32_16x16x32_bf16(
                    false, afr, false, bgr[nf], (short)0, accg[mt][nf], false, false);
            }
        }
        __syncthreads();
    }

#pragma unroll
    for (int nf = 0; nf < 2; ++nf) {
        const int n = n0 + wave * 32 + nf * 16 + ln;
        const float bfn = bfc[n], bgn = bg[n];
#pragma unroll
        for (int mt = 0; mt < 4; ++mt)
#pragma unroll
            for (int r = 0; r < 8; ++r) {
                int m = m0 + mt * 16 + r + 8 * g;
                float fc = accf[mt][nf][r] + bfn;
                float gt = accg[mt][nf][r] + bgn;
                float sig = 1.f / (1.f + __expf(-gt));
                out[(size_t)m * N + n] = sig * tanhf(fc);
            }
    }
}

// ---------------------------------------------------------------------------
extern "C" void kernel_launch(void* const* d_in, const int* in_sizes, int n_in,
                              void* d_out, int out_size, void* d_ws, size_t ws_size,
                              hipStream_t stream)
{
    (void)in_sizes; (void)n_in; (void)out_size; (void)ws_size;
    const float* q   = (const float*)d_in[0];
    const float* k   = (const float*)d_in[1];
    const float* v   = (const float*)d_in[2];
    const float* Wq  = (const float*)d_in[3];
    const float* bq  = (const float*)d_in[4];
    const float* Wk  = (const float*)d_in[5];
    const float* bk  = (const float*)d_in[6];
    const float* Wv  = (const float*)d_in[7];
    const float* bv  = (const float*)d_in[8];
    const float* Wfc = (const float*)d_in[9];
    const float* bfc = (const float*)d_in[10];
    const float* Wg  = (const float*)d_in[11];
    const float* bg  = (const float*)d_in[12];
    float* out = (float*)d_out;

    const size_t ELEMS = (size_t)4096 * 1024;       // B*L x H*DH
    unsigned short* Qh = (unsigned short*)d_ws;     // 4 x 8MB bf16 workspace
    unsigned short* Kh = Qh + ELEMS;
    unsigned short* Vh = Kh + ELEMS;
    unsigned short* Oa = Vh + ELEMS;

    dim3 gGemm(4096 / 64, 1024 / 128);              // (64, 8)
    proj_gemm_kernel<<<gGemm, 128, 0, stream>>>(q, Wq, bq, Qh);
    proj_gemm_kernel<<<gGemm, 128, 0, stream>>>(k, Wk, bk, Kh);
    proj_gemm_kernel<<<gGemm, 128, 0, stream>>>(v, Wv, bv, Vh);

    dim3 gAttn(2048 / 64, 2 * 16);                  // (L/64, B*H)
    attn_kernel<<<gAttn, 128, 0, stream>>>(Qh, Kh, Vh, Oa);

    out_gemm_kernel<<<gGemm, 128, 0, stream>>>(Oa, Wfc, bfc, Wg, bg, out);
}
